// HypBlock_83760452207345
// MI455X (gfx1250) — matrix operand order repaired and checked by reading, not compiled, not run on hardware
//
#include <hip/hip_runtime.h>
#include <cstdint>

// ---------------- types for WMMA ----------------
typedef __attribute__((ext_vector_type(16))) __bf16 v16bf;
typedef __attribute__((ext_vector_type(8)))  float  v8f;
typedef __attribute__((ext_vector_type(4)))  int    v4i;

union Frag {
    uint4 q[2];
    v16bf v;
};
static_assert(sizeof(Frag) == 32, "frag size");

#define S_LEN 197
#define NHEAD 12
#define DMODEL 768
#define DHID 3072
#define NQKV 2304
#define SP 224          // padded score width (multiple of 32, >= 13*16=208)
#define MAXN 0.99999f   // 1 - EPS, EPS = 1e-5

// ---------------- CDNA5 async global->LDS path (guarded, with sync fallback) ---------
#if defined(__has_builtin)
#if __has_builtin(__builtin_amdgcn_global_load_async_to_lds_b128) && \
    __has_builtin(__builtin_amdgcn_s_wait_asynccnt)
#define HAVE_ASYNC_LDS 1
#endif
#endif

#ifdef HAVE_ASYNC_LDS
typedef __attribute__((address_space(1))) v4i gv4i_t;
typedef __attribute__((address_space(3))) v4i lv4i_t;
__device__ __forceinline__ void async_b128(const unsigned short* g, unsigned short* l) {
    // inttoptr casts: global generic pointers carry the 64-bit global address;
    // LDS generic pointers carry the LDS offset in the low 32 bits.
    gv4i_t* gp = (gv4i_t*)(uintptr_t)g;
    lv4i_t* lp = (lv4i_t*)(unsigned int)(uintptr_t)l;
    __builtin_amdgcn_global_load_async_to_lds_b128(gp, lp, 0, 0);
}
#endif

// ---------------- small device helpers ----------------
__device__ __forceinline__ unsigned short f2bf(float f) {
    unsigned int u = __float_as_uint(f);
    unsigned int r = (u + 0x7FFFu + ((u >> 16) & 1u)) >> 16;
    return (unsigned short)r;
}

__device__ __forceinline__ float artanhf_(float x) {
    x = fminf(fmaxf(x, -1.0f + 1e-7f), 1.0f - 1e-7f);
    return 0.5f * (log1pf(x) - log1pf(-x));
}

// block = 256 threads = 8 waves (wave32). Sum-reduce NV values across block.
template <int NV>
__device__ __forceinline__ void blockReduce(float (&v)[NV], float* sm) {
    #pragma unroll
    for (int off = 16; off > 0; off >>= 1) {
        #pragma unroll
        for (int i = 0; i < NV; i++) v[i] += __shfl_xor(v[i], off, 32);
    }
    int lane = threadIdx.x & 31, w = threadIdx.x >> 5;
    if (lane == 0) {
        #pragma unroll
        for (int i = 0; i < NV; i++) sm[w * NV + i] = v[i];
    }
    __syncthreads();
    #pragma unroll
    for (int i = 0; i < NV; i++) {
        float s = 0.f;
        #pragma unroll
        for (int j = 0; j < 8; j++) s += sm[j * NV + i];
        v[i] = s;
    }
    __syncthreads();
}

// ---------------- weight f32 -> bf16 convert ----------------
__global__ void f2bf_kernel(const float* __restrict__ in, unsigned short* __restrict__ out, int n) {
    int i = blockIdx.x * 256 + threadIdx.x;
    if (i < n) out[i] = f2bf(in[i]);
}

// ---------------- K1: xp = project(x); also bf16 copy + |xp|^2 ----------------
__global__ __launch_bounds__(256) void row_project_kernel(
    const float* __restrict__ x, float* __restrict__ xp,
    unsigned short* __restrict__ xpb, float* __restrict__ xpn2)
{
    __shared__ float sm[8];
    int row = blockIdx.x, tid = threadIdx.x;
    const float* xr = x + (size_t)row * DMODEL;
    float v[3], a[1] = {0.f};
    #pragma unroll
    for (int i = 0; i < 3; i++) {
        v[i] = xr[tid + i * 256];
        a[0] += v[i] * v[i];
    }
    blockReduce<1>(a, sm);
    float n = sqrtf(a[0]);
    float s = fminf(1.0f, MAXN / fmaxf(n, 1e-15f));
    #pragma unroll
    for (int i = 0; i < 3; i++) {
        int c = tid + i * 256;
        float o = s * v[i];
        xp[(size_t)row * DMODEL + c]  = o;
        xpb[(size_t)row * DMODEL + c] = f2bf(o);
    }
    if (tid == 0) {
        float nn = n * s;
        xpn2[row] = nn * nn;
    }
}

// ---------------- generic GEMM: C[M,N] = A[M,K] * W[N,K]^T  (bf16 in, f32 out)
// block = 128 threads (4 waves), tile 64x64, K step 32, double-buffered LDS with
// async global->LDS staging when available.
__global__ __launch_bounds__(128) void gemm_bf16_tn(
    const unsigned short* __restrict__ A, const unsigned short* __restrict__ W,
    float* __restrict__ C, int M, int N, int K)
{
    __shared__ __align__(16) unsigned short As[2][64 * 40];  // row stride 40 bf16 = 80B
    __shared__ __align__(16) unsigned short Bs[2][64 * 40];

    int tid  = threadIdx.x;
    int lane = tid & 31, w = tid >> 5;
    int bm = blockIdx.y * 64, bn = blockIdx.x * 64;
    int l15 = lane & 15, kh = lane >> 4;

    v8f acc[4] = {};

    int lrow = tid >> 1, lhalf = tid & 1;  // 64 rows x 2 halves of 16 cols
    int ar = bm + lrow;
    int br = bn + lrow;
    int nk = K >> 5;

#ifdef HAVE_ASYNC_LDS
    // --- async double-buffered staging ---
    {
        // prologue: tile 0 into buffer 0
        const unsigned short* ag = A + (size_t)ar * K + lhalf * 16;
        if (ar < M) {
            unsigned short* al = &As[0][lrow * 40 + lhalf * 16];
            async_b128(ag, al);
            async_b128(ag + 8, al + 8);
        }
        const unsigned short* bg = W + (size_t)br * K + lhalf * 16;
        unsigned short* bl = &Bs[0][lrow * 40 + lhalf * 16];
        async_b128(bg, bl);
        async_b128(bg + 8, bl + 8);
    }
    for (int i = 0; i < nk; i++) {
        int cur = i & 1;
        if (i + 1 < nk) {
            int kb = (i + 1) << 5;
            const unsigned short* ag = A + (size_t)ar * K + kb + lhalf * 16;
            if (ar < M) {
                unsigned short* al = &As[cur ^ 1][lrow * 40 + lhalf * 16];
                async_b128(ag, al);
                async_b128(ag + 8, al + 8);
            }
            const unsigned short* bg = W + (size_t)br * K + kb + lhalf * 16;
            unsigned short* bl = &Bs[cur ^ 1][lrow * 40 + lhalf * 16];
            async_b128(bg, bl);
            async_b128(bg + 8, bl + 8);
            __builtin_amdgcn_s_wait_asynccnt(4);  // leave only the just-issued tile
        } else {
            __builtin_amdgcn_s_wait_asynccnt(0);
        }
        __syncthreads();  // tile `cur` resident for all waves

        Frag af;
        const char* arow = (const char*)(&As[cur][(w * 16 + l15) * 40]);
        af.q[0] = *(const uint4*)(arow + kh * 32);        // K 16*kh + 0..7
        af.q[1] = *(const uint4*)(arow + kh * 32 + 16);   // K 16*kh + 8..15
        #pragma unroll
        for (int t = 0; t < 4; t++) {
            Frag bf_;
            const char* brow = (const char*)(&Bs[cur][(t * 16 + l15) * 40]);
            bf_.q[0] = *(const uint4*)(brow + kh * 32);        // K 16*kh + 0..7
            bf_.q[1] = *(const uint4*)(brow + kh * 32 + 16);   // K 16*kh + 8..15
            acc[t] = __builtin_amdgcn_wmma_f32_16x16x32_bf16(
                false, af.v, false, bf_.v, (short)0, acc[t], false, false);
        }
        __syncthreads();  // everyone done reading before buffer is overwritten
    }
#else
    // --- synchronous fallback staging ---
    for (int kb = 0; kb < K; kb += 32) {
        uint4 a0 = {0, 0, 0, 0}, a1 = {0, 0, 0, 0};
        if (ar < M) {
            const uint4* ap = (const uint4*)(A + (size_t)ar * K + kb + lhalf * 16);
            a0 = ap[0]; a1 = ap[1];
        }
        uint4* asp = (uint4*)(&As[0][lrow * 40 + lhalf * 16]);
        asp[0] = a0; asp[1] = a1;
        const uint4* bp = (const uint4*)(W + (size_t)br * K + kb + lhalf * 16);
        uint4 b0 = bp[0], b1 = bp[1];
        uint4* bsp = (uint4*)(&Bs[0][lrow * 40 + lhalf * 16]);
        bsp[0] = b0; bsp[1] = b1;
        __syncthreads();

        Frag af;
        const char* arow = (const char*)(&As[0][(w * 16 + l15) * 40]);
        af.q[0] = *(const uint4*)(arow + kh * 32);
        af.q[1] = *(const uint4*)(arow + kh * 32 + 16);
        #pragma unroll
        for (int t = 0; t < 4; t++) {
            Frag bf_;
            const char* brow = (const char*)(&Bs[0][(t * 16 + l15) * 40]);
            bf_.q[0] = *(const uint4*)(brow + kh * 32);
            bf_.q[1] = *(const uint4*)(brow + kh * 32 + 16);
            acc[t] = __builtin_amdgcn_wmma_f32_16x16x32_bf16(
                false, af.v, false, bf_.v, (short)0, acc[t], false, false);
        }
        __syncthreads();
    }
#endif

    #pragma unroll
    for (int t = 0; t < 4; t++) {
        int n = bn + t * 16 + l15;
        #pragma unroll
        for (int j = 0; j < 8; j++) {
            int m = bm + w * 16 + 8 * kh + j;
            if (m < M) C[(size_t)m * N + n] = acc[t][j];
        }
    }
}

// ---------------- K3: split qkv, expmap0+project on q,k, bf16 all, qq/kk ----------------
__global__ __launch_bounds__(256) void qkv_split_kernel(
    const float* __restrict__ qkv,
    unsigned short* __restrict__ qb, unsigned short* __restrict__ kb,
    unsigned short* __restrict__ vb,
    float* __restrict__ qq, float* __restrict__ kk)
{
    int row = blockIdx.x;                 // b*S + s
    int b = row / S_LEN, s = row % S_LEN;
    int lane = threadIdx.x & 31, w = threadIdx.x >> 5;
    const float* rp = qkv + (size_t)row * NQKV;

    for (int hv = w; hv < 36; hv += 8) {
        int type = hv / NHEAD, h = hv % NHEAD;
        const float* v = rp + type * DMODEL + h * 64;
        float v0 = v[lane], v1 = v[lane + 32];
        size_t obase = ((size_t)(b * NHEAD + h) * S_LEN + s) * 64;
        if (type == 2) {
            vb[obase + lane]      = f2bf(v0);
            vb[obase + lane + 32] = f2bf(v1);
        } else {
            float ss = v0 * v0 + v1 * v1;
            #pragma unroll
            for (int off = 16; off > 0; off >>= 1) ss += __shfl_xor(ss, off, 32);
            float n  = sqrtf(ss);
            float se = tanhf(n) / fmaxf(n, 1e-15f);   // expmap0 scale
            float pn = se * n;                        // norm after expmap0
            float sp = fminf(1.0f, MAXN / fmaxf(pn, 1e-15f));
            float sc = se * sp;
            float nn = pn * sp;
            unsigned short* ob = (type == 0) ? qb : kb;
            ob[obase + lane]      = f2bf(sc * v0);
            ob[obase + lane + 32] = f2bf(sc * v1);
            if (lane == 0) {
                float* oq = (type == 0) ? qq : kk;
                oq[(size_t)(b * NHEAD + h) * S_LEN + s] = nn * nn;
            }
        }
    }
}

// ---------------- K4: Poincare-distance attention, one (head, 16-q-row) tile/block -----
__global__ __launch_bounds__(256) void attn_kernel(
    const unsigned short* __restrict__ qb, const unsigned short* __restrict__ kb,
    const unsigned short* __restrict__ vb,
    const float* __restrict__ qq, const float* __restrict__ kk,
    unsigned short* __restrict__ aout /* [M, DMODEL] bf16 */)
{
    __shared__ __align__(16) unsigned short qs[16 * 72];     // 16x64 q tile, stride 72
    __shared__ float  scores[16 * SP];
    __shared__ __align__(16) unsigned short abuf[16 * SP];   // softmaxed attn bf16
    __shared__ __align__(16) unsigned short vT[64 * 232];    // transposed V, stride 232
    __shared__ float qqs[16];
    __shared__ float kks[SP];

    int bh = blockIdx.y;
    int b = bh / NHEAD, h = bh % NHEAD;
    int qi0 = blockIdx.x * 16;
    int tid = threadIdx.x, lane = tid & 31, w = tid >> 5;
    int l15 = lane & 15, kh = lane >> 4;

    const unsigned short* qhead = qb + (size_t)bh * S_LEN * 64;
    const unsigned short* khead = kb + (size_t)bh * S_LEN * 64;
    const unsigned short* vhead = vb + (size_t)bh * S_LEN * 64;

    // stage q tile (zero beyond S)
    if (tid < 128) {
        int r = tid >> 3, c8 = (tid & 7) * 8;
        uint4 val = {0, 0, 0, 0};
        int srow = qi0 + r;
        if (srow < S_LEN) val = *(const uint4*)(qhead + srow * 64 + c8);
        *(uint4*)(qs + r * 72 + c8) = val;
    }
    if (tid < 16) qqs[tid] = (qi0 + tid < S_LEN) ? qq[(size_t)bh * S_LEN + qi0 + tid] : 0.f;
    for (int c = tid; c < SP; c += 256) kks[c] = (c < S_LEN) ? kk[(size_t)bh * S_LEN + c] : 0.f;
    {   // init score pad cols 208..223 to -inf
        int r = tid >> 4, c = 208 + (tid & 15);
        scores[r * SP + c] = -3.0e38f;
    }
    // stage V transposed (zero beyond S)
    for (int idx = tid; idx < SP * 64; idx += 256) {
        int s = idx >> 6, dd = idx & 63;
        unsigned short vv = 0;
        if (s < S_LEN) vv = vhead[s * 64 + dd];
        vT[dd * 232 + s] = vv;
    }
    __syncthreads();

    // ---- QK^T via WMMA, then Poincare distance -> scores ----
    for (int tc = w; tc < 13; tc += 8) {
        v8f acc = {};
        int n = tc * 16 + l15;
        const unsigned short* krow = khead + (size_t)n * 64;
        #pragma unroll
        for (int ks = 0; ks < 64; ks += 32) {
            Frag af, bf_;
            const char* arow = (const char*)(qs + l15 * 72) + ks * 2;
            af.q[0] = *(const uint4*)(arow + kh * 32);
            af.q[1] = *(const uint4*)(arow + kh * 32 + 16);
            const unsigned short* kp = krow + ks + kh * 16;
            bf_.q[0] = *(const uint4*)(kp);
            bf_.q[1] = *(const uint4*)(kp + 8);
            acc = __builtin_amdgcn_wmma_f32_16x16x32_bf16(
                false, af.v, false, bf_.v, (short)0, acc, false, false);
        }
        float kkv = kks[n];
        #pragma unroll
        for (int j = 0; j < 8; j++) {
            int m = j + 8 * kh;
            float qqv = qqs[m];
            float qkv = acc[j];
            float num = qqv - 2.f * qkv + kkv;
            float den = fmaxf(1.f - 2.f * qkv + qqv * kkv, 1e-15f);
            float r2  = fminf(fmaxf(num / den, 1e-15f), 1.f - 1e-7f);
            float dist = 2.f * artanhf_(sqrtf(r2));
            float sc = -dist * 0.125f;  // scale = 64^-0.5
            if (n >= S_LEN) sc = -3.0e38f;
            scores[m * SP + n] = sc;
        }
    }
    __syncthreads();

    // ---- softmax (wave w handles rows 2w, 2w+1) ----
    for (int r = 2 * w; r < 2 * w + 2; r++) {
        float mx = -3.4e38f;
        for (int c = lane; c < SP; c += 32) mx = fmaxf(mx, scores[r * SP + c]);
        #pragma unroll
        for (int off = 16; off > 0; off >>= 1) mx = fmaxf(mx, __shfl_xor(mx, off, 32));
        float sum = 0.f;
        for (int c = lane; c < SP; c += 32) {
            float e = __expf(scores[r * SP + c] - mx);
            scores[r * SP + c] = e;
            sum += e;
        }
        #pragma unroll
        for (int off = 16; off > 0; off >>= 1) sum += __shfl_xor(sum, off, 32);
        float inv = 1.f / fmaxf(sum, 1e-30f);
        for (int c = lane; c < SP; c += 32) abuf[r * SP + c] = f2bf(scores[r * SP + c] * inv);
    }
    __syncthreads();

    // ---- attn @ V via WMMA (waves 0..3, one 16-wide d-tile each) ----
    if (w < 4) {
        v8f acc = {};
        int dd = w * 16 + l15;
        #pragma unroll
        for (int ks = 0; ks < SP; ks += 32) {
            Frag af, bf_;
            const char* arow = (const char*)(abuf + l15 * SP) + ks * 2;
            af.q[0] = *(const uint4*)(arow + kh * 32);
            af.q[1] = *(const uint4*)(arow + kh * 32 + 16);
            const char* brow = (const char*)(vT + dd * 232) + ks * 2;
            bf_.q[0] = *(const uint4*)(brow + kh * 32);
            bf_.q[1] = *(const uint4*)(brow + kh * 32 + 16);
            acc = __builtin_amdgcn_wmma_f32_16x16x32_bf16(
                false, af.v, false, bf_.v, (short)0, acc, false, false);
        }
        #pragma unroll
        for (int j = 0; j < 8; j++) {
            int srow = qi0 + j + 8 * kh;
            if (srow < S_LEN) {
                size_t o = ((size_t)b * S_LEN + srow) * DMODEL + h * 64 + dd;
                aout[o] = f2bf(acc[j]);
            }
        }
    }
}

// ---------------- K5: y = project(expmap0(proj+b)); x1 = project(mobius(xp,y)) --------
__global__ __launch_bounds__(256) void resid1_kernel(
    const float* __restrict__ pout, const float* __restrict__ bproj,
    const float* __restrict__ xp, const float* __restrict__ xpn2,
    float* __restrict__ x1, unsigned short* __restrict__ x1b, float* __restrict__ x1n)
{
    __shared__ float sm[8 * 2];
    int row = blockIdx.x, tid = threadIdx.x;
    const float* pr = pout + (size_t)row * DMODEL;
    const float* xr = xp + (size_t)row * DMODEL;
    float tv[3], xv[3], a[2] = {0.f, 0.f};
    #pragma unroll
    for (int i = 0; i < 3; i++) {
        int c = tid + i * 256;
        tv[i] = pr[c] + bproj[c];
        xv[i] = xr[c];
        a[0] += tv[i] * tv[i];
        a[1] += xv[i] * tv[i];
    }
    blockReduce<2>(a, sm);
    float tn  = sqrtf(a[0]);
    float s1  = tanhf(tn) / fmaxf(tn, 1e-15f);          // expmap0
    float ynr = s1 * tn;
    float s2  = fminf(1.f, MAXN / fmaxf(ynr, 1e-15f));  // project
    float sy  = s1 * s2;
    float yn  = ynr * s2, y2 = yn * yn;
    float x2  = xpn2[row];
    float xy  = sy * a[1];
    float den = fmaxf(1.f + 2.f * xy + x2 * y2, 1e-15f);
    float A = (1.f + 2.f * xy + y2) / den, B = (1.f - x2) / den;
    float n1 = sqrtf(fmaxf(A * A * x2 + 2.f * A * B * xy + B * B * y2, 0.f));
    float s3 = fminf(1.f, MAXN / fmaxf(n1, 1e-15f));
    #pragma unroll
    for (int i = 0; i < 3; i++) {
        int c = tid + i * 256;
        float o = s3 * (A * xv[i] + B * sy * tv[i]);
        x1[(size_t)row * DMODEL + c]  = o;
        x1b[(size_t)row * DMODEL + c] = f2bf(o);
    }
    if (tid == 0) x1n[row] = n1 * s3;
}

// ---------------- K6a: h = hyp_linear epilogue (width 3072), bf16 out + |h| -----------
__global__ __launch_bounds__(256) void hyp1_kernel(
    const float* __restrict__ mx, const float* __restrict__ bvec,
    const float* __restrict__ xnorm,
    unsigned short* __restrict__ hb, float* __restrict__ hn)
{
    __shared__ float sm[8 * 3];
    int row = blockIdx.x, tid = threadIdx.x;
    const float* mr = mx + (size_t)row * DHID;
    float mv[12], bv[12], a[3] = {0.f, 0.f, 0.f};
    #pragma unroll
    for (int i = 0; i < 12; i++) {
        int c = tid + i * 256;
        mv[i] = mr[c];
        bv[i] = bvec[c];
        a[0] += mv[i] * mv[i];
        a[1] += mv[i] * bv[i];
        a[2] += bv[i] * bv[i];
    }
    blockReduce<3>(a, sm);
    float mxn = fmaxf(sqrtf(a[0]), 1e-15f);
    float xn  = fmaxf(xnorm[row], 1e-15f);
    float th  = tanhf(mxn / xn * artanhf_(xn));  // mobius_matvec magnitude
    float smv = th / mxn;
    float h0n = fabsf(th);
    float sp  = fminf(1.f, MAXN / fmaxf(h0n, 1e-15f));  // project
    smv *= sp; h0n *= sp;
    float x2 = h0n * h0n, y2 = a[2];
    float xy = smv * a[1];
    float den = fmaxf(1.f + 2.f * xy + x2 * y2, 1e-15f);
    float A = (1.f + 2.f * xy + y2) / den, B = (1.f - x2) / den;
    float hn2 = fmaxf(A * A * x2 + 2.f * A * B * xy + B * B * y2, 0.f);
    float hnn = sqrtf(hn2);
    float s3  = fminf(1.f, MAXN / fmaxf(hnn, 1e-15f));
    float al = s3 * A * smv, be = s3 * B;
    #pragma unroll
    for (int i = 0; i < 12; i++) {
        int c = tid + i * 256;
        hb[(size_t)row * DHID + c] = f2bf(al * mv[i] + be * bv[i]);
    }
    if (tid == 0) hn[row] = hnn * s3;
}

// ---------------- K6b: h2 epilogue (width 768) + final mobius residual -> d_out -------
__global__ __launch_bounds__(256) void final_kernel(
    const float* __restrict__ mx2, const float* __restrict__ b2,
    const float* __restrict__ hnorm,
    const float* __restrict__ x1, const float* __restrict__ x1n,
    float* __restrict__ out)
{
    __shared__ float sm[8 * 5];
    int row = blockIdx.x, tid = threadIdx.x;
    const float* mr = mx2 + (size_t)row * DMODEL;
    const float* xr = x1 + (size_t)row * DMODEL;
    float mv[3], bv[3], xv[3], a[5] = {0.f, 0.f, 0.f, 0.f, 0.f};
    #pragma unroll
    for (int i = 0; i < 3; i++) {
        int c = tid + i * 256;
        mv[i] = mr[c]; bv[i] = b2[c]; xv[i] = xr[c];
        a[0] += mv[i] * mv[i];
        a[1] += mv[i] * bv[i];
        a[2] += bv[i] * bv[i];
        a[3] += xv[i] * mv[i];
        a[4] += xv[i] * bv[i];
    }
    blockReduce<5>(a, sm);
    // hyp_linear(h, W2, b2) scalar chain
    float mxn = fmaxf(sqrtf(a[0]), 1e-15f);
    float xn  = fmaxf(hnorm[row], 1e-15f);
    float th  = tanhf(mxn / xn * artanhf_(xn));
    float smv = th / mxn;
    float h0n = fabsf(th);
    float sp  = fminf(1.f, MAXN / fmaxf(h0n, 1e-15f));
    smv *= sp; h0n *= sp;
    float x2 = h0n * h0n, y2 = a[2];
    float xy = smv * a[1];
    float den = fmaxf(1.f + 2.f * xy + x2 * y2, 1e-15f);
    float A1 = (1.f + 2.f * xy + y2) / den, B1 = (1.f - x2) / den;
    float h2n2 = fmaxf(A1 * A1 * x2 + 2.f * A1 * B1 * xy + B1 * B1 * y2, 0.f);
    float h2n  = sqrtf(h2n2);
    float s3   = fminf(1.f, MAXN / fmaxf(h2n, 1e-15f));
    float al = s3 * A1 * smv, be = s3 * B1;
    h2n *= s3; h2n2 = h2n * h2n;
    // final: project(mobius_add(x1, h2))
    float xf2 = x1n[row]; xf2 = xf2 * xf2;
    float xyf = al * a[3] + be * a[4];
    float denf = fmaxf(1.f + 2.f * xyf + xf2 * h2n2, 1e-15f);
    float Af = (1.f + 2.f * xyf + h2n2) / denf, Bf = (1.f - xf2) / denf;
    float on2 = fmaxf(Af * Af * xf2 + 2.f * Af * Bf * xyf + Bf * Bf * h2n2, 0.f);
    float so  = fminf(1.f, MAXN / fmaxf(sqrtf(on2), 1e-15f));
    #pragma unroll
    for (int i = 0; i < 3; i++) {
        int c = tid + i * 256;
        out[(size_t)row * DMODEL + c] = so * (Af * xv[i] + Bf * (al * mv[i] + be * bv[i]));
    }
}

// ---------------- host-side orchestration ----------------
extern "C" void kernel_launch(void* const* d_in, const int* in_sizes, int n_in,
                              void* d_out, int out_size, void* d_ws, size_t ws_size,
                              hipStream_t stream) {
    (void)in_sizes; (void)n_in; (void)out_size; (void)ws_size;
    const float* x     = (const float*)d_in[0];
    const float* Wqkv  = (const float*)d_in[1];
    const float* Wproj = (const float*)d_in[2];
    const float* bproj = (const float*)d_in[3];
    const float* W1    = (const float*)d_in[4];
    const float* b1    = (const float*)d_in[5];
    const float* W2    = (const float*)d_in[6];
    const float* b2    = (const float*)d_in[7];
    float* out = (float*)d_out;

    const int M = 32 * S_LEN;  // 6304

    // deterministic workspace arena
    char* ws = (char*)d_ws;
    size_t off = 0;
    auto carve = [&](size_t bytes) -> char* {
        char* p = ws + off;
        off = (off + bytes + 255) & ~(size_t)255;
        return p;
    };
    unsigned short* wqkvb  = (unsigned short*)carve((size_t)NQKV * DMODEL * 2);
    unsigned short* wprojb = (unsigned short*)carve((size_t)DMODEL * DMODEL * 2);
    unsigned short* w1b    = (unsigned short*)carve((size_t)DHID * DMODEL * 2);
    unsigned short* w2b    = (unsigned short*)carve((size_t)DMODEL * DHID * 2);
    float* P0 = (float*)carve((size_t)M * DHID * 4);    // qkv / mx / mx2
    float* P1 = (float*)carve((size_t)M * DMODEL * 4);  // xp
    float* P3 = (float*)carve((size_t)M * DMODEL * 4);  // x1
    unsigned short* Q0 = (unsigned short*)carve(((size_t)M * DHID + 8192) * 2);  // bf16 staging
    float* qq   = (float*)carve((size_t)384 * S_LEN * 4 + 256);
    float* kk   = (float*)carve((size_t)384 * S_LEN * 4 + 256);
    float* xpn2 = (float*)carve((size_t)M * 4);
    float* x1n  = (float*)carve((size_t)M * 4);
    float* hn   = (float*)carve((size_t)M * 4);

    unsigned short* s0 = Q0;                               // xp_bf / qb / x1_bf
    unsigned short* s1 = Q0 + (size_t)M * DMODEL;          // kb
    unsigned short* s2 = Q0 + (size_t)2 * M * DMODEL;      // vb
    unsigned short* s3 = Q0 + (size_t)3 * M * DMODEL;      // attn out bf16

    // weight conversions
    {
        int n;
        n = NQKV * DMODEL;  f2bf_kernel<<<(n + 255) / 256, 256, 0, stream>>>(Wqkv, wqkvb, n);
        n = DMODEL * DMODEL;f2bf_kernel<<<(n + 255) / 256, 256, 0, stream>>>(Wproj, wprojb, n);
        n = DHID * DMODEL;  f2bf_kernel<<<(n + 255) / 256, 256, 0, stream>>>(W1, w1b, n);
        n = DMODEL * DHID;  f2bf_kernel<<<(n + 255) / 256, 256, 0, stream>>>(W2, w2b, n);
    }

    // 1) xp = project(x)
    row_project_kernel<<<M, 256, 0, stream>>>(x, P1, s0, xpn2);

    // 2) qkv = xp @ Wqkv^T
    {
        dim3 g(NQKV / 64, (M + 63) / 64);
        gemm_bf16_tn<<<g, 128, 0, stream>>>(s0, wqkvb, P0, M, NQKV, DMODEL);
    }

    // 3) split + expmap0/project q,k
    qkv_split_kernel<<<M, 256, 0, stream>>>(P0, s0, s1, s2, qq, kk);

    // 4) attention
    {
        dim3 g(13, 32 * NHEAD);
        attn_kernel<<<g, 256, 0, stream>>>(s0, s1, s2, qq, kk, s3);
    }

    // 5) proj GEMM
    {
        dim3 g(DMODEL / 64, (M + 63) / 64);
        gemm_bf16_tn<<<g, 128, 0, stream>>>(s3, wprojb, P0, M, DMODEL, DMODEL);
    }

    // 6) residual 1: x1 = project(mobius(xp, project(expmap0(proj + b))))
    resid1_kernel<<<M, 256, 0, stream>>>(P0, bproj, P1, xpn2, P3, s0, x1n);

    // 7) mx = x1 @ W1^T
    {
        dim3 g(DHID / 64, (M + 63) / 64);
        gemm_bf16_tn<<<g, 128, 0, stream>>>(s0, w1b, P0, M, DHID, DMODEL);
    }

    // 8) h = hyp_linear epilogue (write bf16 row-major [M, 3072] into Q0)
    hyp1_kernel<<<M, 256, 0, stream>>>(P0, b1, x1n, Q0, hn);

    // 9) mx2 = h @ W2^T
    {
        dim3 g(DMODEL / 64, (M + 63) / 64);
        gemm_bf16_tn<<<g, 128, 0, stream>>>(Q0, w2b, P0, M, DMODEL, DHID);
    }

    // 10) h2 epilogue + final mobius residual
    final_kernel<<<M, 256, 0, stream>>>(P0, b2, hn, P3, x1n, out);
}
